// HASTModel_89395449299821
// MI455X (gfx1250) — compile-verified, matrix operand order checked by reading, not run on hardware
//
#include <hip/hip_runtime.h>
#include <hip/hip_bf16.h>
#include <math.h>

// ---------------------------------------------------------------------------
// HAST forward for MI455X (gfx1250, wave32, WMMA).
// All GEMM-shaped work routed through v_wmma_f32_16x16x32_f16 (f32 accum).
// Per-(n,t) 64x64 tiles live entirely in LDS; model state fits in L2.
// ---------------------------------------------------------------------------

typedef __attribute__((ext_vector_type(16))) _Float16 v16h;
typedef __attribute__((ext_vector_type(8)))  float    v8f;

#define NB     4
#define BBANDS 5
#define CCH    64
#define TSEQ   512
#define DSP    64
#define DTM    128
#define DFFS   256
#define DFFT   512
#define DEMB   64
#define NHS    4
#define NHT    4
#define WINW   10

__device__ __forceinline__ int lane_id() { return (int)(threadIdx.x & 31u); }

// K slot for half-element i (0..15) in a 16x32 f16 A/B fragment (ISA 7.12.2).
__device__ __forceinline__ int frag_k(int i, int lane) {
  return ((i & 8) ? 16 : 0) + ((lane & 16) ? 8 : 0) + (((i >> 1) & 3) << 1) + (i & 1);
}

// A fragment (16x32) from f16 LDS, row-major [M][lda]
__device__ __forceinline__ v16h frag_a_lds(const _Float16* A, int lda, int row0, int k0) {
  int lane = lane_id(); int m = lane & 15; v16h r;
#pragma unroll
  for (int i = 0; i < 16; ++i) r[i] = A[(row0 + m) * lda + k0 + frag_k(i, lane)];
  return r;
}
// A fragment with zero-padding beyond kmax (for dh=16 heads padded to K=32)
__device__ __forceinline__ v16h frag_a_lds_kmax(const _Float16* A, int lda, int row0, int k0, int kmax) {
  int lane = lane_id(); int m = lane & 15; v16h r;
#pragma unroll
  for (int i = 0; i < 16; ++i) {
    int k = frag_k(i, lane);
    r[i] = (k < kmax) ? A[(row0 + m) * lda + k0 + k] : (_Float16)0.0f;
  }
  return r;
}
// A fragment from f32 LDS (convert on load)
__device__ __forceinline__ v16h frag_a_lds32(const float* A, int lda, int row0, int k0) {
  int lane = lane_id(); int m = lane & 15; v16h r;
#pragma unroll
  for (int i = 0; i < 16; ++i) r[i] = (_Float16)A[(row0 + m) * lda + k0 + frag_k(i, lane)];
  return r;
}
// A fragment gathered from f32 global
__device__ __forceinline__ v16h frag_a_f32g(const float* A, int lda, int row0, int k0) {
  int lane = lane_id(); int m = lane & 15; v16h r;
#pragma unroll
  for (int i = 0; i < 16; ++i) r[i] = (_Float16)A[(size_t)(row0 + m) * lda + k0 + frag_k(i, lane)];
  return r;
}
// B fragment (32x16) from f16 LDS, row-major [K][ldb]
__device__ __forceinline__ v16h frag_b_lds(const _Float16* B, int ldb, int k0, int col0) {
  int lane = lane_id(); int nn = lane & 15; v16h r;
#pragma unroll
  for (int i = 0; i < 16; ++i) r[i] = B[(k0 + frag_k(i, lane)) * ldb + col0 + nn];
  return r;
}
// B fragment gathered from f32 global, row-major [K][ldb]
__device__ __forceinline__ v16h frag_b_f32g(const float* B, int ldb, int k0, int col0) {
  int lane = lane_id(); int nn = lane & 15; v16h r;
#pragma unroll
  for (int i = 0; i < 16; ++i) r[i] = (_Float16)B[(size_t)(k0 + frag_k(i, lane)) * ldb + col0 + nn];
  return r;
}

__device__ __forceinline__ v8f wmma16(v16h a, v16h b, v8f c) {
  return __builtin_amdgcn_wmma_f32_16x16x32_f16(false, a, false, b, (short)0, c, false, false);
}

// C-tile store into f32 LDS, row-major [M][ldc]
__device__ __forceinline__ void store_c_lds(float* Cm, int ldc, int row0, int col0, const v8f& acc) {
  int lane = lane_id(); int rb = (lane & 16) ? 8 : 0; int cb = lane & 15;
#pragma unroll
  for (int i = 0; i < 8; ++i) Cm[(row0 + rb + i) * ldc + col0 + cb] = acc[i];
}

__device__ __forceinline__ float sigmoidf_(float x) { return 1.0f / (1.0f + __expf(-x)); }

// ---------------------------------------------------------------------------
// 1) u[n,t,b] = log1p(cumsum_t(mean_c x^2)/(t+1))
// ---------------------------------------------------------------------------
__global__ void k_band_u(const float* __restrict__ x, float* __restrict__ u) {
  int nb = blockIdx.x;                 // n*BBANDS + b
  int t  = threadIdx.x;                // 0..511
  __shared__ float sc[TSEQ];
  const float* xp = x + (size_t)nb * CCH * TSEQ;
  float s = 0.f;
  for (int c = 0; c < CCH; ++c) { float v = xp[c * TSEQ + t]; s += v * v; }
  sc[t] = s * (1.0f / CCH);
  __syncthreads();
  for (int off = 1; off < TSEQ; off <<= 1) {
    float v = sc[t];
    float add = (t >= off) ? sc[t - off] : 0.f;
    __syncthreads();
    sc[t] = v + add;
    __syncthreads();
  }
  float cm = sc[t] / (float)(t + 1);
  int n = nb / BBANDS, b = nb % BBANDS;
  u[((size_t)n * TSEQ + t) * BBANDS + b] = log1pf(cm);
}

// ---------------------------------------------------------------------------
// 2) alpha = softmax(relu(u@W1+b1)@W2+b2); top-3 renormalized weights
// ---------------------------------------------------------------------------
__global__ void k_alpha(const float* __restrict__ u,
                        const float* __restrict__ W1, const float* __restrict__ b1,
                        const float* __restrict__ W2, const float* __restrict__ b2,
                        float* __restrict__ alpha_out, float* __restrict__ ren) {
  int idx = blockIdx.x * blockDim.x + threadIdx.x;   // n*TSEQ + t
  if (idx >= NB * TSEQ) return;
  float uv[BBANDS], logit[BBANDS];
  for (int b = 0; b < BBANDS; ++b) uv[b] = u[(size_t)idx * BBANDS + b];
  for (int b = 0; b < BBANDS; ++b) logit[b] = b2[b];
  for (int j = 0; j < 32; ++j) {
    float hj = b1[j];
    for (int b = 0; b < BBANDS; ++b) hj += uv[b] * W1[b * 32 + j];
    hj = fmaxf(hj, 0.f);
    for (int b = 0; b < BBANDS; ++b) logit[b] += hj * W2[j * BBANDS + b];
  }
  float mx = -1e30f;
  for (int b = 0; b < BBANDS; ++b) mx = fmaxf(mx, logit[b]);
  float sum = 0.f, a[BBANDS];
  for (int b = 0; b < BBANDS; ++b) { a[b] = __expf(logit[b] - mx); sum += a[b]; }
  float inv = 1.f / sum;
  for (int b = 0; b < BBANDS; ++b) { a[b] *= inv; alpha_out[(size_t)idx * BBANDS + b] = a[b]; }
  int used[BBANDS] = {0, 0, 0, 0, 0};
  for (int k = 0; k < 3; ++k) {
    int bi = 0; float bv = -1e30f;
    for (int b = 0; b < BBANDS; ++b) if (!used[b] && a[b] > bv) { bv = a[b]; bi = b; }
    used[bi] = 1;
  }
  float ms = 0.f;
  for (int b = 0; b < BBANDS; ++b) if (used[b]) ms += a[b];
  ms = fmaxf(ms, 1e-8f);
  for (int b = 0; b < BBANDS; ++b) ren[(size_t)idx * BBANDS + b] = used[b] ? a[b] / ms : 0.f;
}

// ---------------------------------------------------------------------------
// 3) X_fused[n,c,t] = sum_b alpha[n,t,b] * x[n,b,c,t]
// ---------------------------------------------------------------------------
__global__ void k_xfused(const float* __restrict__ x, const float* __restrict__ alpha,
                         float* __restrict__ Xf) {
  int idx = blockIdx.x * blockDim.x + threadIdx.x;
  if (idx >= NB * CCH * TSEQ) return;
  int t = idx % TSEQ; int c = (idx / TSEQ) % CCH; int n = idx / (TSEQ * CCH);
  float s = 0.f;
  for (int b = 0; b < BBANDS; ++b)
    s += alpha[((size_t)n * TSEQ + t) * BBANDS + b] * x[(((size_t)n * BBANDS + b) * CCH + c) * TSEQ + t];
  Xf[idx] = s;
}

// ---------------------------------------------------------------------------
// 4) Emean[n,b,t,d] = mean_c E[n,b,c,t,d]   (E = relu(x*W1+b1)@W2+b2)
// ---------------------------------------------------------------------------
__global__ void k_emean(const float* __restrict__ x,
                        const float* __restrict__ W1, const float* __restrict__ b1,
                        const float* __restrict__ W2, const float* __restrict__ b2,
                        float* __restrict__ Emean) {
  int blk = blockIdx.x;                // (n*BBANDS+b)*TSEQ + t
  int t = blk % TSEQ; int nb = blk / TSEQ;
  __shared__ float hl[CCH][16];
  __shared__ float hs[16];
  int tid = threadIdx.x;               // 0..63
  float xv = x[((size_t)nb * CCH + tid) * TSEQ + t];
  for (int k = 0; k < 16; ++k) hl[tid][k] = fmaxf(xv * W1[k] + b1[k], 0.f);
  __syncthreads();
  if (tid < 16) { float s = 0.f; for (int c = 0; c < CCH; ++c) s += hl[c][tid]; hs[tid] = s * (1.f / CCH); }
  __syncthreads();
  float e = b2[tid];
  for (int k = 0; k < 16; ++k) e += hs[k] * W2[k * DEMB + tid];
  Emean[(size_t)blk * DEMB + tid] = e;
}

// ---------------------------------------------------------------------------
// 5) g[n,t] = sigmoid(embed_mean @ gw + gb)
// ---------------------------------------------------------------------------
__global__ void k_gate(const float* __restrict__ ren, const float* __restrict__ Emean,
                       const float* __restrict__ gw, const float* __restrict__ gb,
                       float* __restrict__ g_out) {
  int idx = blockIdx.x * blockDim.x + threadIdx.x;   // n*TSEQ + t
  if (idx >= NB * TSEQ) return;
  int t = idx % TSEQ; int n = idx / TSEQ;
  float acc = gb[0];
  for (int d = 0; d < DEMB; ++d) {
    float em = 0.f;
    for (int b = 0; b < BBANDS; ++b)
      em += ren[(size_t)idx * BBANDS + b] * Emean[(((size_t)n * BBANDS + b) * TSEQ + t) * DEMB + d];
    acc += em * gw[d];
  }
  g_out[idx] = sigmoidf_(acc);
}

// ---------------------------------------------------------------------------
// 6) P_func / P_t: per (n,t), for each band compute E (on-the-fly, LDS f16),
//    S = softmax(E E^T / 8) via WMMA, accumulate ren-weighted.
// ---------------------------------------------------------------------------
__global__ void __launch_bounds__(128) k_pfunc(
    const float* __restrict__ x,
    const float* __restrict__ fpW1, const float* __restrict__ fpb1,
    const float* __restrict__ fpW2, const float* __restrict__ fpb2,
    const float* __restrict__ ren, const float* __restrict__ gv_,
    float* __restrict__ pf_out, float* __restrict__ pt_out) {
  int idx = blockIdx.x;               // n*TSEQ + t
  int t = idx % TSEQ; int n = idx / TSEQ;
  __shared__ _Float16 E16[CCH * DEMB];
  __shared__ float Sm[CCH * CCH];
  __shared__ float Pacc[CCH * CCH];
  int tid = threadIdx.x, w = tid >> 5;
  for (int i = tid; i < CCH * CCH; i += 128) Pacc[i] = 0.f;
  __syncthreads();
  for (int b = 0; b < BBANDS; ++b) {
    if (tid < CCH) {
      float xv = x[(((size_t)n * BBANDS + b) * CCH + tid) * TSEQ + t];
      float hh[16];
      for (int k = 0; k < 16; ++k) hh[k] = fmaxf(xv * fpW1[k] + fpb1[k], 0.f);
      for (int d = 0; d < DEMB; ++d) {
        float e = fpb2[d];
        for (int k = 0; k < 16; ++k) e += hh[k] * fpW2[k * DEMB + d];
        E16[tid * DEMB + d] = (_Float16)e;
      }
    }
    __syncthreads();
#pragma unroll
    for (int tj = 0; tj < 4; ++tj) {
      v8f acc = {};
      for (int k0 = 0; k0 < DEMB; k0 += 32) {
        v16h a  = frag_a_lds(E16, DEMB, w * 16, k0);
        v16h bb = frag_a_lds(E16, DEMB, tj * 16, k0);   // B = E^T
        acc = wmma16(a, bb, acc);
      }
#pragma unroll
      for (int i = 0; i < 8; ++i) acc[i] *= 0.125f;     // 1/sqrt(64)
      store_c_lds(Sm, CCH, w * 16, tj * 16, acc);
    }
    __syncthreads();
    if (tid < CCH) {
      float mx = -1e30f;
      for (int j = 0; j < CCH; ++j) mx = fmaxf(mx, Sm[tid * CCH + j]);
      float s = 0.f;
      for (int j = 0; j < CCH; ++j) { float e = __expf(Sm[tid * CCH + j] - mx); s += e; Sm[tid * CCH + j] = e; }
      float inv = 1.f / s;
      for (int j = 0; j < CCH; ++j) Sm[tid * CCH + j] *= inv;
    }
    __syncthreads();
    float rb = ren[(size_t)idx * BBANDS + b];
    for (int i = tid; i < CCH * CCH; i += 128) Pacc[i] += rb * Sm[i];
    __syncthreads();
  }
  float gv = gv_[idx];
  size_t base = (size_t)idx * CCH * CCH;
  for (int i = tid; i < CCH * CCH; i += 128) {
    float pf = Pacc[i];
    pf_out[base + i] = pf;
    pt_out[base + i] = gv * (1.f / CCH) + (1.f - gv) * pf;
  }
}

// ---------------------------------------------------------------------------
// 7) h[n,t,c,:] = Xf[n,c,t]*in_W + in_b + pos[c,:]
// ---------------------------------------------------------------------------
__global__ void k_embed(const float* __restrict__ Xf, const float* __restrict__ inW,
                        const float* __restrict__ inb, const float* __restrict__ pos,
                        float* __restrict__ h) {
  size_t idx = (size_t)blockIdx.x * blockDim.x + threadIdx.x;
  if (idx >= (size_t)NB * TSEQ * CCH * DSP) return;
  int d = idx % DSP; int c = (idx / DSP) % CCH;
  int t = (idx / ((size_t)DSP * CCH)) % TSEQ; int n = idx / ((size_t)DSP * CCH * TSEQ);
  h[idx] = Xf[((size_t)n * CCH + c) * TSEQ + t] * inW[d] + inb[d] + pos[c * DSP + d];
}

// ---------------------------------------------------------------------------
// 8) Spatial attention, fused per (n,t): QKV proj + 4-head attn (+P_t bias)
//    + O-proj.  All WMMA, operands in LDS.
//    NOTE: no brace-initialized arrays of LDS pointers (lld addrspacecast bug);
//    QKV lives in one flat LDS buffer, indexed at runtime.
// ---------------------------------------------------------------------------
__global__ void __launch_bounds__(128) k_sattn(
    const float* __restrict__ h,
    const float* __restrict__ Wq, const float* __restrict__ bq,
    const float* __restrict__ Wk, const float* __restrict__ bk,
    const float* __restrict__ Wv, const float* __restrict__ bv,
    const float* __restrict__ Wo, const float* __restrict__ bo,
    const float* __restrict__ Pt, float* __restrict__ outp) {
  int idx = blockIdx.x;               // n*TSEQ + t
  __shared__ _Float16 h16[CCH * DSP];
  __shared__ _Float16 qkv16[3 * CCH * DSP];   // q | k | v
  __shared__ float    Sc[CCH * CCH];
  __shared__ _Float16 ctx16[CCH * DSP];
  int tid = threadIdx.x, w = tid >> 5;
  int lane = tid & 31, rb = (lane & 16) ? 8 : 0, cb = lane & 15;
  const float* hp = h + (size_t)idx * CCH * DSP;
  for (int i = tid; i < CCH * DSP; i += 128) h16[i] = (_Float16)hp[i];
  __syncthreads();
  // QKV projections (weights gathered straight from global; shared L2-resident)
  for (int p = 0; p < 3; ++p) {
    const float* Wp = (p == 0) ? Wq : ((p == 1) ? Wk : Wv);
    const float* bp = (p == 0) ? bq : ((p == 1) ? bk : bv);
    _Float16* dst = qkv16 + p * (CCH * DSP);
    v8f acc[4] = {};
    for (int k0 = 0; k0 < DSP; k0 += 32) {
      v16h a = frag_a_lds(h16, DSP, w * 16, k0);
#pragma unroll
      for (int tj = 0; tj < 4; ++tj) {
        v16h bb = frag_b_f32g(Wp, DSP, k0, tj * 16);
        acc[tj] = wmma16(a, bb, acc[tj]);
      }
    }
#pragma unroll
    for (int tj = 0; tj < 4; ++tj) {
      int col = tj * 16 + cb;
#pragma unroll
      for (int i = 0; i < 8; ++i)
        dst[(w * 16 + rb + i) * DSP + col] = (_Float16)(acc[tj][i] + bp[col]);
    }
  }
  __syncthreads();
  const _Float16* q16  = qkv16;
  const _Float16* kk16 = qkv16 + CCH * DSP;
  const _Float16* vv16 = qkv16 + 2 * CCH * DSP;
  // per-head attention
  const float* ptp = Pt + (size_t)idx * CCH * CCH;
  for (int hh = 0; hh < NHS; ++hh) {
    v16h a = frag_a_lds_kmax(q16, DSP, w * 16, hh * 16, 16);
#pragma unroll
    for (int tj = 0; tj < 4; ++tj) {
      v16h bb = frag_a_lds_kmax(kk16, DSP, tj * 16, hh * 16, 16);  // B = K^T
      v8f acc = {};
      acc = wmma16(a, bb, acc);
#pragma unroll
      for (int i = 0; i < 8; ++i) {
        int row = w * 16 + rb + i, col = tj * 16 + cb;
        Sc[row * CCH + col] = acc[i] * 0.25f + ptp[(size_t)row * CCH + col];
      }
    }
    __syncthreads();
    if (tid < CCH) {
      float mx = -1e30f;
      for (int j = 0; j < CCH; ++j) mx = fmaxf(mx, Sc[tid * CCH + j]);
      float s = 0.f;
      for (int j = 0; j < CCH; ++j) { float e = __expf(Sc[tid * CCH + j] - mx); s += e; Sc[tid * CCH + j] = e; }
      float inv = 1.f / s;
      for (int j = 0; j < CCH; ++j) Sc[tid * CCH + j] *= inv;
    }
    __syncthreads();
    {
      v8f acc = {};
      for (int k0 = 0; k0 < CCH; k0 += 32) {
        v16h aa = frag_a_lds32(Sc, CCH, w * 16, k0);
        v16h bb = frag_b_lds(vv16, DSP, k0, hh * 16);
        acc = wmma16(aa, bb, acc);
      }
      int col = hh * 16 + cb;
#pragma unroll
      for (int i = 0; i < 8; ++i) ctx16[(w * 16 + rb + i) * DSP + col] = (_Float16)acc[i];
    }
    __syncthreads();
  }
  // O projection -> global
  float* op = outp + (size_t)idx * CCH * DSP;
  v8f acc[4] = {};
  for (int k0 = 0; k0 < DSP; k0 += 32) {
    v16h a = frag_a_lds(ctx16, DSP, w * 16, k0);
#pragma unroll
    for (int tj = 0; tj < 4; ++tj) {
      v16h bb = frag_b_f32g(Wo, DSP, k0, tj * 16);
      acc[tj] = wmma16(a, bb, acc[tj]);
    }
  }
#pragma unroll
  for (int tj = 0; tj < 4; ++tj) {
    int col = tj * 16 + cb;
#pragma unroll
    for (int i = 0; i < 8; ++i)
      op[(size_t)(w * 16 + rb + i) * DSP + col] = acc[tj][i] + bo[col];
  }
}

// ---------------------------------------------------------------------------
// 9) h = LN(h + o) per row of width D
// ---------------------------------------------------------------------------
__global__ void k_addln(float* __restrict__ h, const float* __restrict__ o,
                        const float* __restrict__ s, const float* __restrict__ b,
                        int M, int D) {
  int r = blockIdx.x * blockDim.x + threadIdx.x;
  if (r >= M) return;
  float* hr = h + (size_t)r * D;
  const float* orr = o + (size_t)r * D;
  float mean = 0.f;
  for (int d = 0; d < D; ++d) mean += hr[d] + orr[d];
  mean /= (float)D;
  float var = 0.f;
  for (int d = 0; d < D; ++d) { float v = hr[d] + orr[d] - mean; var += v * v; }
  var /= (float)D;
  float inv = rsqrtf(var + 1e-5f);
  for (int d = 0; d < D; ++d) {
    float v = hr[d] + orr[d];
    hr[d] = (v - mean) * inv * s[d] + b[d];
  }
}

// ---------------------------------------------------------------------------
// 10) Fused FFN: out = relu(h@F1+f1)@F2+f2, 64-row panel per block,
//     hidden streamed through LDS in 64-col chunks; out accumulators in VGPRs.
// ---------------------------------------------------------------------------
template <int D, int DFF>
__global__ void __launch_bounds__(128) k_ffn(
    const float* __restrict__ h,
    const float* __restrict__ F1, const float* __restrict__ f1,
    const float* __restrict__ F2, const float* __restrict__ f2,
    float* __restrict__ outp) {
  int row0 = blockIdx.x * 64;
  __shared__ _Float16 a16[64 * D];
  __shared__ _Float16 hc[64 * 64];
  int tid = threadIdx.x, w = tid >> 5;
  int lane = tid & 31, rb = (lane & 16) ? 8 : 0, cb = lane & 15;
  for (int i = tid; i < 64 * D; i += 128) a16[i] = (_Float16)h[(size_t)row0 * D + i];
  __syncthreads();
  v8f oacc[D / 16];
#pragma unroll
  for (int tc = 0; tc < D / 16; ++tc) oacc[tc] = (v8f){};
  for (int nc = 0; nc < DFF; nc += 64) {
    v8f acc[4] = {};
#pragma unroll
    for (int k0 = 0; k0 < D; k0 += 32) {
      v16h a = frag_a_lds(a16, D, w * 16, k0);
#pragma unroll
      for (int tj = 0; tj < 4; ++tj) {
        v16h bb = frag_b_f32g(F1, DFF, k0, nc + tj * 16);
        acc[tj] = wmma16(a, bb, acc[tj]);
      }
    }
    __syncthreads();                   // hc from previous chunk fully consumed
#pragma unroll
    for (int tj = 0; tj < 4; ++tj) {
      int col = nc + tj * 16 + cb;
#pragma unroll
      for (int i = 0; i < 8; ++i)
        hc[(w * 16 + rb + i) * 64 + tj * 16 + cb] = (_Float16)fmaxf(acc[tj][i] + f1[col], 0.f);
    }
    __syncthreads();
#pragma unroll
    for (int k0 = 0; k0 < 64; k0 += 32) {
      v16h a = frag_a_lds(hc, 64, w * 16, k0);
#pragma unroll
      for (int tc = 0; tc < D / 16; ++tc) {
        v16h bb = frag_b_f32g(F2, D, nc + k0, tc * 16);
        oacc[tc] = wmma16(a, bb, oacc[tc]);
      }
    }
  }
#pragma unroll
  for (int tc = 0; tc < D / 16; ++tc) {
    int col = tc * 16 + cb;
#pragma unroll
    for (int i = 0; i < 8; ++i)
      outp[(size_t)(row0 + w * 16 + rb + i) * D + col] = oacc[tc][i] + f2[col];
  }
}

// ---------------------------------------------------------------------------
// 11) Generic GEMM: C = [relu](A@B + bias), 64x64 tile per block
// ---------------------------------------------------------------------------
__global__ void __launch_bounds__(128) k_gemm(
    const float* __restrict__ A, const float* __restrict__ B,
    const float* __restrict__ bias, float* __restrict__ Cout,
    int M, int Nn, int K, int relu) {
  int row0 = blockIdx.x * 64, col0 = blockIdx.y * 64;
  __shared__ _Float16 a16[64 * 32];
  __shared__ _Float16 b16[32 * 64];
  int tid = threadIdx.x, w = tid >> 5;
  int lane = tid & 31, rb = (lane & 16) ? 8 : 0, cb = lane & 15;
  v8f acc[4] = {};
  for (int k0 = 0; k0 < K; k0 += 32) {
    __syncthreads();
    for (int i = tid; i < 64 * 32; i += 128) {
      int r = i >> 5, c = i & 31;
      a16[i] = (_Float16)A[(size_t)(row0 + r) * K + k0 + c];
    }
    for (int i = tid; i < 32 * 64; i += 128) {
      int r = i >> 6, c = i & 63;
      b16[i] = (_Float16)B[(size_t)(k0 + r) * Nn + col0 + c];
    }
    if (k0 + 32 < K) {                 // CDNA5 global_prefetch_b8 of next panels
      __builtin_prefetch(&A[(size_t)(row0 + (tid & 63)) * K + k0 + 32], 0, 1);
      if (tid < 32) __builtin_prefetch(&B[(size_t)(k0 + 32 + tid) * Nn + col0], 0, 1);
    }
    __syncthreads();
    v16h a = frag_a_lds(a16, 32, w * 16, 0);
#pragma unroll
    for (int tj = 0; tj < 4; ++tj) {
      v16h bb = frag_b_lds(b16, 64, 0, tj * 16);
      acc[tj] = wmma16(a, bb, acc[tj]);
    }
  }
#pragma unroll
  for (int tj = 0; tj < 4; ++tj) {
    int col = col0 + tj * 16 + cb;
#pragma unroll
    for (int i = 0; i < 8; ++i) {
      float v = acc[tj][i] + (bias ? bias[col] : 0.f);
      if (relu) v = fmaxf(v, 0.f);
      Cout[(size_t)(row0 + w * 16 + rb + i) * Nn + col] = v;
    }
  }
}

// ---------------------------------------------------------------------------
// 12) Temporal windowed attention: per (n, head, 16-row i-tile), only the two
//     diagonal-band j-tiles are computed (exact: exp(-1e9) underflows to 0).
// ---------------------------------------------------------------------------
__global__ void __launch_bounds__(32) k_tattn(
    const float* __restrict__ q, const float* __restrict__ kM,
    const float* __restrict__ v, float* __restrict__ ctx) {
  const int numTi = TSEQ / 16;
  int blk = blockIdx.x;
  int ti = blk % numTi; int nh = blk / numTi;
  int hh = nh % NHT; int n = nh / NHT;
  int i0 = ti * 16;
  int j0 = (ti > 0) ? (ti - 1) * 16 : 0;
  const float* qp = q  + (size_t)n * TSEQ * DTM;
  const float* kp = kM + (size_t)n * TSEQ * DTM;
  const float* vp = v  + (size_t)n * TSEQ * DTM;
  __shared__ float    sc[16 * 32];
  __shared__ _Float16 p16[16 * 32];
  int lane = threadIdx.x, rb = (lane & 16) ? 8 : 0, cb = lane & 15;
  const float scale = 0.17677669529663687f;   // 1/sqrt(32)
  v16h a = frag_a_f32g(qp + (size_t)i0 * DTM + hh * 32, DTM, 0, 0);
#pragma unroll
  for (int tj = 0; tj < 2; ++tj) {
    v16h bb = frag_a_f32g(kp + (size_t)(j0 + tj * 16) * DTM + hh * 32, DTM, 0, 0);  // K^T
    v8f acc = {};
    acc = wmma16(a, bb, acc);
#pragma unroll
    for (int i = 0; i < 8; ++i) {
      int irow = rb + i, gi = i0 + irow, gj = j0 + tj * 16 + cb;
      float s = acc[i] * scale;
      if (gj > gi || (gi - gj) >= WINW) s = -1e9f;
      sc[irow * 32 + tj * 16 + cb] = s;
    }
  }
  __syncthreads();
  if (lane < 16) {
    float mx = -1e30f;
    for (int j = 0; j < 32; ++j) mx = fmaxf(mx, sc[lane * 32 + j]);
    float s = 0.f;
    for (int j = 0; j < 32; ++j) { float e = __expf(sc[lane * 32 + j] - mx); s += e; sc[lane * 32 + j] = e; }
    float inv = 1.f / s;
    for (int j = 0; j < 32; ++j) p16[lane * 32 + j] = (_Float16)(sc[lane * 32 + j] * inv);
  }
  __syncthreads();
  v16h pa = frag_a_lds(p16, 32, 0, 0);
#pragma unroll
  for (int nt = 0; nt < 2; ++nt) {
    v16h bb = frag_b_f32g(vp + (size_t)j0 * DTM + hh * 32 + nt * 16, DTM, 0, 0);
    v8f acc = {};
    acc = wmma16(pa, bb, acc);
#pragma unroll
    for (int i = 0; i < 8; ++i)
      ctx[(size_t)n * TSEQ * DTM + (size_t)(i0 + rb + i) * DTM + hh * 32 + nt * 16 + cb] = acc[i];
  }
}

// ---------------------------------------------------------------------------
// 13) head: y_seq = sigmoid(z @ hw + hb); y_final = y_seq[:, -1]
// ---------------------------------------------------------------------------
__global__ void k_head(const float* __restrict__ z, const float* __restrict__ hw,
                       const float* __restrict__ hb, float* __restrict__ yfin,
                       float* __restrict__ yseq) {
  int idx = blockIdx.x * blockDim.x + threadIdx.x;   // n*TSEQ + t
  if (idx >= NB * TSEQ) return;
  float s = hb[0];
  for (int d = 0; d < DTM; ++d) s += z[(size_t)idx * DTM + d] * hw[d];
  float y = sigmoidf_(s);
  yseq[idx] = y;
  if ((idx % TSEQ) == TSEQ - 1) yfin[idx / TSEQ] = y;
}

// ---------------------------------------------------------------------------
extern "C" void kernel_launch(void* const* d_in, const int* in_sizes, int n_in,
                              void* d_out, int out_size, void* d_ws, size_t ws_size,
                              hipStream_t stream) {
  (void)in_sizes; (void)n_in; (void)out_size; (void)ws_size;
  auto F = [&](int i) { return (const float*)d_in[i]; };
  const float* x = F(0);

  // workspace layout (floats)
  float* W = (float*)d_ws;
  size_t o = 0;
  float* u     = W + o; o += (size_t)NB * TSEQ * BBANDS;
  float* ren   = W + o; o += (size_t)NB * TSEQ * BBANDS;
  float* Xf    = W + o; o += (size_t)NB * CCH * TSEQ;
  float* Emean = W + o; o += (size_t)NB * BBANDS * TSEQ * DEMB;
  float* h     = W + o; o += (size_t)NB * TSEQ * CCH * DSP;
  float* tmp1  = W + o; o += (size_t)NB * TSEQ * CCH * DSP;
  float* z     = W + o; o += (size_t)NB * TSEQ * DTM;
  float* zq    = W + o; o += (size_t)NB * TSEQ * DTM;
  float* zk    = W + o; o += (size_t)NB * TSEQ * DTM;
  float* zv    = W + o; o += (size_t)NB * TSEQ * DTM;
  float* zctx  = W + o; o += (size_t)NB * TSEQ * DTM;
  float* ztmp  = W + o; o += (size_t)NB * TSEQ * DTM;

  // output layout: (y_final, y_seq, alpha, P_func, g, P_t)
  float* out      = (float*)d_out;
  float* y_final  = out;
  float* y_seq    = out + NB;
  float* alpha_o  = y_seq + (size_t)NB * TSEQ;
  float* pf_o     = alpha_o + (size_t)NB * TSEQ * BBANDS;
  float* g_o      = pf_o + (size_t)NB * TSEQ * CCH * CCH;
  float* pt_o     = g_o + (size_t)NB * TSEQ;

  const int NT = NB * TSEQ;                 // 2048
  const int MROWS = NB * TSEQ * CCH;        // 131072

  k_band_u<<<NB * BBANDS, TSEQ, 0, stream>>>(x, u);
  k_alpha<<<(NT + 255) / 256, 256, 0, stream>>>(u, F(1), F(2), F(3), F(4), alpha_o, ren);
  k_xfused<<<(NB * CCH * TSEQ + 255) / 256, 256, 0, stream>>>(x, alpha_o, Xf);
  k_emean<<<NB * BBANDS * TSEQ, 64, 0, stream>>>(x, F(5), F(6), F(7), F(8), Emean);
  k_gate<<<(NT + 255) / 256, 256, 0, stream>>>(ren, Emean, F(9), F(10), g_o);
  k_pfunc<<<NT, 128, 0, stream>>>(x, F(5), F(6), F(7), F(8), ren, g_o, pf_o, pt_o);
  k_embed<<<(MROWS * DSP + 255) / 256, 256, 0, stream>>>(Xf, F(11), F(12), F(13), h);

  // spatial transformer, 2 layers
  for (int L = 0; L < 2; ++L) {
    int p = 14 + L * 16;
    k_sattn<<<NT, 128, 0, stream>>>(h, F(p + 0), F(p + 1), F(p + 2), F(p + 3),
                                    F(p + 4), F(p + 5), F(p + 6), F(p + 7), pt_o, tmp1);
    k_addln<<<(MROWS + 255) / 256, 256, 0, stream>>>(h, tmp1, F(p + 8), F(p + 9), MROWS, DSP);
    k_ffn<DSP, DFFS><<<MROWS / 64, 128, 0, stream>>>(h, F(p + 10), F(p + 11), F(p + 12), F(p + 13), tmp1);
    k_addln<<<(MROWS + 255) / 256, 256, 0, stream>>>(h, tmp1, F(p + 14), F(p + 15), MROWS, DSP);
  }

  // temporal projection: (n*T, C*Ds) @ (4096,128)
  k_gemm<<<dim3(NT / 64, DTM / 64), 128, 0, stream>>>(h, F(46), F(47), z, NT, DTM, CCH * DSP, 0);

  // temporal layer
  {
    int p = 48;
    k_gemm<<<dim3(NT / 64, DTM / 64), 128, 0, stream>>>(z, F(p + 0), F(p + 1), zq, NT, DTM, DTM, 0);
    k_gemm<<<dim3(NT / 64, DTM / 64), 128, 0, stream>>>(z, F(p + 2), F(p + 3), zk, NT, DTM, DTM, 0);
    k_gemm<<<dim3(NT / 64, DTM / 64), 128, 0, stream>>>(z, F(p + 4), F(p + 5), zv, NT, DTM, DTM, 0);
    k_tattn<<<NB * NHT * (TSEQ / 16), 32, 0, stream>>>(zq, zk, zv, zctx);
    k_gemm<<<dim3(NT / 64, DTM / 64), 128, 0, stream>>>(zctx, F(p + 6), F(p + 7), ztmp, NT, DTM, DTM, 0);
    k_addln<<<(NT + 255) / 256, 256, 0, stream>>>(z, ztmp, F(p + 8), F(p + 9), NT, DTM);
    k_ffn<DTM, DFFT><<<NT / 64, 128, 0, stream>>>(z, F(p + 10), F(p + 11), F(p + 12), F(p + 13), ztmp);
    k_addln<<<(NT + 255) / 256, 256, 0, stream>>>(z, ztmp, F(p + 14), F(p + 15), NT, DTM);
  }

  k_head<<<(NT + 255) / 256, 256, 0, stream>>>(z, F(64), F(65), y_final, y_seq);
}
